// CausalSelfAttention_32598801776812
// MI455X (gfx1250) — matrix operand order repaired and checked by reading, not compiled, not run on hardware
//
#include <hip/hip_runtime.h>
#include <stdint.h>

// ---------------------------------------------------------------------------
// CDNA5 (gfx1250) causal self-attention with RoPE.
// Matmuls: v_wmma_f32_16x16x32_bf16 (bf16 A/B, f32 accumulate).
// Attention K/V staging: global_load_async_to_lds_b128 (ASYNCcnt-tracked),
// double-buffered.  GEMM: double-buffered LDS + register prefetch.
// ---------------------------------------------------------------------------

typedef __attribute__((ext_vector_type(16))) __bf16 v16bf;
typedef __attribute__((ext_vector_type(8)))  float  v8f;

#define DIM      1024
#define HEADS    16
#define HEAD_DIM 64
#define SEQ      2048
#define BATCH    4

// ---- bf16 helpers (storage type: unsigned short) --------------------------
__device__ __forceinline__ unsigned short f2bf(float f) {
    unsigned int u = __builtin_bit_cast(unsigned int, f);
    u = u + 0x7fffu + ((u >> 16) & 1u);           // round-to-nearest-even
    return (unsigned short)(u >> 16);
}
__device__ __forceinline__ float bf2f(unsigned short h) {
    unsigned int u = ((unsigned int)h) << 16;
    return __builtin_bit_cast(float, u);
}

struct U4Pair { uint4 a, b; };
// Build a 16x32 bf16 WMMA fragment from two 16-byte aligned 8-element chunks.
__device__ __forceinline__ v16bf frag_from(const unsigned short* p0,
                                           const unsigned short* p1) {
    U4Pair p;
    p.a = *(const uint4*)p0;
    p.b = *(const uint4*)p1;
    return __builtin_bit_cast(v16bf, p);
}

__device__ __forceinline__ v8f wmma_bf16(v16bf a, v16bf b, v8f c) {
    return __builtin_amdgcn_wmma_f32_16x16x32_bf16(
        /*neg_a=*/false, a, /*neg_b=*/false, b,
        /*c_mod=*/(short)0, c, /*reuse_a=*/false, /*reuse_b=*/false);
}

// ---- CDNA5 async global->LDS copy (16B), IOFFSET applies to both sides ----
#define ASYNC_CP16(lds32, gaddr64, OFF)                                        \
    asm volatile("global_load_async_to_lds_b128 %0, %1, off offset:" #OFF      \
                 :: "v"(lds32), "v"(gaddr64) : "memory")
#define WAIT_ASYNC(N) asm volatile("s_wait_asynccnt " #N ::: "memory")

// ===========================================================================
// 1) Tiled GEMM: C[M,N] = A[M,K] * B[K,N], A/B fp32 in HBM, bf16 in WMMA.
//    Block tile 128x128x32, 256 threads = 8 waves, each wave 32(M)x64(N).
//    Double-buffered LDS, register prefetch of tile k+1 over WMMAs on k.
// ===========================================================================
#define LDS_A 40   // 32 + 8 pad (elements), row stride 80B (16B multiple)
#define LDS_B 40

template <bool OUT_BF16>
__global__ __launch_bounds__(256)
void gemm_wmma_bf16(const float* __restrict__ A, const float* __restrict__ B,
                    void* __restrict__ Cout, int M, int N, int K) {
    __shared__ unsigned short sA[2][128 * LDS_A];   // [row][k]
    __shared__ unsigned short sB[2][128 * LDS_B];   // [n][k]

    const int tid  = threadIdx.x;
    const int lane = tid & 31;
    const int wave = tid >> 5;
    const int mW   = (wave & 3) * 32;   // 4 waves along M
    const int nW   = (wave >> 2) * 64;  // 2 waves along N
    const int bm   = blockIdx.y * 128;
    const int bn   = blockIdx.x * 128;
    const int l15  = lane & 15;
    const int hi   = lane >> 4;

    // staging roles
    const int ar = tid >> 1, ac = (tid & 1) * 16;   // A: 2 thr/row, 16 elems
    const int bk = tid >> 3, bc = (tid & 7) * 16;   // B: 8 thr/row, 16 elems
    const float* aSrc = A + (size_t)(bm + ar) * K + ac;
    const float* bSrc = B + (size_t)bk * N + bn + bc;

    float4 aReg[4], bReg[4];
    auto gload = [&](int k0) {
        #pragma unroll
        for (int q = 0; q < 4; ++q) {
            aReg[q] = *(const float4*)(aSrc + k0 + q * 4);
            bReg[q] = *(const float4*)(bSrc + (size_t)k0 * N + q * 4);
        }
    };
    auto lstore = [&](int buf) {
        unsigned short* da = &sA[buf][ar * LDS_A + ac];
        #pragma unroll
        for (int q = 0; q < 4; ++q) {
            da[q * 4 + 0] = f2bf(aReg[q].x);
            da[q * 4 + 1] = f2bf(aReg[q].y);
            da[q * 4 + 2] = f2bf(aReg[q].z);
            da[q * 4 + 3] = f2bf(aReg[q].w);
            sB[buf][(bc + q * 4 + 0) * LDS_B + bk] = f2bf(bReg[q].x);
            sB[buf][(bc + q * 4 + 1) * LDS_B + bk] = f2bf(bReg[q].y);
            sB[buf][(bc + q * 4 + 2) * LDS_B + bk] = f2bf(bReg[q].z);
            sB[buf][(bc + q * 4 + 3) * LDS_B + bk] = f2bf(bReg[q].w);
        }
    };

    v8f acc[2][4];
    #pragma unroll
    for (int i = 0; i < 2; ++i)
        #pragma unroll
        for (int j = 0; j < 4; ++j)
            for (int e = 0; e < 8; ++e) acc[i][j][e] = 0.0f;

    gload(0);
    lstore(0);
    __syncthreads();

    for (int k0 = 0; k0 < K; k0 += 32) {
        const int  cur = (k0 >> 5) & 1;
        const bool nx  = (k0 + 32) < K;
        if (nx) gload(k0 + 32);                 // prefetch over the WMMAs

        v16bf afrag[2], bfrag[4];
        #pragma unroll
        for (int i = 0; i < 2; ++i) {
            const unsigned short* base = &sA[cur][(mW + i * 16 + l15) * LDS_A];
            afrag[i] = frag_from(base + hi * 8, base + 16 + hi * 8);
        }
        #pragma unroll
        for (int j = 0; j < 4; ++j) {
            const unsigned short* base =
                &sB[cur][(nW + j * 16 + l15) * LDS_B];
            bfrag[j] = frag_from(base + hi * 8, base + 16 + hi * 8);
        }
        #pragma unroll
        for (int i = 0; i < 2; ++i)
            #pragma unroll
            for (int j = 0; j < 4; ++j)
                acc[i][j] = wmma_bf16(afrag[i], bfrag[j], acc[i][j]);

        if (nx) lstore(cur ^ 1);
        __syncthreads();
    }

    // ---- epilogue: lane(l15)=N, vgpr r = M (+8 for lanes 16-31) --------
    #pragma unroll
    for (int i = 0; i < 2; ++i) {
        #pragma unroll
        for (int j = 0; j < 4; ++j) {
            const size_t base =
                (size_t)(bm + mW + i * 16 + hi * 8) * N + (bn + nW + j * 16 + l15);
            #pragma unroll
            for (int r = 0; r < 8; ++r) {
                const float v = acc[i][j][r];
                if (OUT_BF16)
                    ((unsigned short*)Cout)[base + (size_t)r * N] = f2bf(v);
                else
                    ((float*)Cout)[base + (size_t)r * N] = v;
            }
        }
    }
}

// ===========================================================================
// 2) RoPE + head reshape.  qkv bf16 [B*T, 3*DIM] ->
//    q,k bf16 [B,H,T,64], v transposed bf16 [B,H,64,T].
// ===========================================================================
__global__ __launch_bounds__(256)
void rope_reshape(const unsigned short* __restrict__ qkv,
                  unsigned short* __restrict__ qo,
                  unsigned short* __restrict__ ko,
                  unsigned short* __restrict__ vTo) {
    const size_t i = (size_t)blockIdx.x * 256 + threadIdx.x;  // B*T*DIM threads
    const int c  = (int)(i & (DIM - 1));
    const size_t bt = i >> 10;
    const int b  = (int)(bt >> 11);
    const int t  = (int)(bt & (SEQ - 1));
    const int h  = c >> 6;
    const int d  = c & 63;

    const unsigned short* row = qkv + bt * (3 * DIM);
    const float qv = bf2f(row[c]);
    const float kv = bf2f(row[DIM + c]);
    const float vv = bf2f(row[2 * DIM + c]);
    const float qp = bf2f(row[c ^ 32]);           // rotate-half partner
    const float kp = bf2f(row[DIM + (c ^ 32)]);

    const int dh = d & 31;
    // freq = theta^(-2*dh/64) = exp(-dh * ln(10000)/32)
    const float freq = __expf(-(float)dh * 0.28782313662425572f);
    const float ang  = (float)t * freq;
    float sn, cs;
    __sincosf(ang, &sn, &cs);

    const float qrot = (d < 32) ? -qp : qp;
    const float krot = (d < 32) ? -kp : kp;
    const float qr = qv * cs + qrot * sn;
    const float kr = kv * cs + krot * sn;

    const size_t bh = (size_t)b * HEADS + h;
    qo [(bh * SEQ + t) * HEAD_DIM + d] = f2bf(qr);
    ko [(bh * SEQ + t) * HEAD_DIM + d] = f2bf(kr);
    vTo[(bh * HEAD_DIM + d) * SEQ + t] = f2bf(vv);
}

// ===========================================================================
// 3) Flash attention (causal, online softmax).
//    grid(T/64, B*H), 128 threads = 4 waves; each wave owns 16 Q rows.
//    K/V tiles double-buffered via async global->LDS copies (ASYNCcnt).
// ===========================================================================
#define LDS_KV 72   // 64 + 8 pad; row stride 144B (16B multiple)

__global__ __launch_bounds__(128)
void flash_attn(const unsigned short* __restrict__ q,
                const unsigned short* __restrict__ k,
                const unsigned short* __restrict__ vT,
                float* __restrict__ o) {
    __shared__ unsigned short sK[2][64 * LDS_KV];     // [key][d]
    __shared__ unsigned short sV[2][64 * LDS_KV];     // [d][key]
    __shared__ unsigned short sP[4 * 16 * LDS_KV];    // per-wave P stage

    const int tid   = threadIdx.x;
    const int lane  = tid & 31;
    const int wave  = tid >> 5;
    const int l15   = lane & 15;
    const int hi    = lane >> 4;
    const int bh    = blockIdx.y;
    const int b     = bh >> 4;
    const int h     = bh & 15;
    const int qbase = blockIdx.x * 64;
    const int qrow0 = qbase + wave * 16;
    const int rowoff = hi * 8;

    // Issue one K tile + one V^T tile as 8 async 16B copies per thread-pair.
    const int sr = tid >> 1;            // 0..63 (key row / d row)
    const int scc = (tid & 1) * 32;     // element offset 0 / 32
    auto issue_tile = [&](int buf, int kbase) {
        unsigned long long ga = (unsigned long long)(uintptr_t)
            (k + ((size_t)bh * SEQ + kbase + sr) * HEAD_DIM + scc);
        unsigned long long gv = (unsigned long long)(uintptr_t)
            (vT + ((size_t)bh * HEAD_DIM + sr) * SEQ + kbase + scc);
        unsigned lk = (unsigned)(uintptr_t)&sK[buf][sr * LDS_KV + scc];
        unsigned lv = (unsigned)(uintptr_t)&sV[buf][sr * LDS_KV + scc];
        ASYNC_CP16(lk, ga, 0);  ASYNC_CP16(lk, ga, 16);
        ASYNC_CP16(lk, ga, 32); ASYNC_CP16(lk, ga, 48);
        ASYNC_CP16(lv, gv, 0);  ASYNC_CP16(lv, gv, 16);
        ASYNC_CP16(lv, gv, 32); ASYNC_CP16(lv, gv, 48);
    };

    // Q fragments live in registers for the whole kernel (K-dim = d: 0..63).
    const unsigned short* qrow =
        q + ((size_t)bh * SEQ + qrow0 + l15) * HEAD_DIM;
    v16bf qf[2];
    #pragma unroll
    for (int i = 0; i < 2; ++i)
        qf[i] = frag_from(qrow + i * 32 + hi * 8, qrow + i * 32 + 16 + hi * 8);

    float mrow[8], lrow[8];
    v8f oacc[4];
    #pragma unroll
    for (int r = 0; r < 8; ++r) { mrow[r] = -1e30f; lrow[r] = 0.0f; }
    #pragma unroll
    for (int j = 0; j < 4; ++j)
        for (int e = 0; e < 8; ++e) oacc[j][e] = 0.0f;

    unsigned short* pw = &sP[wave * 16 * LDS_KV];
    const int ntiles = blockIdx.x + 1;        // causal: tiles up to diagonal

    issue_tile(0, 0);                         // prologue: tile 0 in flight

    for (int kt = 0; kt < ntiles; ++kt) {
        const int cur   = kt & 1;
        const int kbase = kt * 64;
        if (kt + 1 < ntiles) {
            issue_tile(cur ^ 1, kbase + 64);  // prefetch next tile
            WAIT_ASYNC(0x8);                  // tile kt resident (in-order)
        } else {
            WAIT_ASYNC(0x0);
        }
        __syncthreads();

        // ---- S = Q K^T  (4 key-subtiles, K-dim split 0..31 / 32..63) ----
        v8f s[4];
        #pragma unroll
        for (int j = 0; j < 4; ++j) {
            for (int e = 0; e < 8; ++e) s[j][e] = 0.0f;
            const unsigned short* base = &sK[cur][(j * 16 + l15) * LDS_KV];
            v16bf b0 = frag_from(base + hi * 8,      base + 16 + hi * 8);
            v16bf b1 = frag_from(base + 32 + hi * 8, base + 48 + hi * 8);
            s[j] = wmma_bf16(qf[0], b0, s[j]);
            s[j] = wmma_bf16(qf[1], b1, s[j]);
        }

        // ---- scale + causal mask ---------------------------------------
        const float sc = 0.125f;   // 1/sqrt(64)
        #pragma unroll
        for (int j = 0; j < 4; ++j) {
            const int col = kbase + j * 16 + l15;
            #pragma unroll
            for (int r = 0; r < 8; ++r) {
                float v = s[j][r] * sc;
                if (col > qrow0 + r + rowoff) v = -1e30f;
                s[j][r] = v;
            }
        }

        // ---- online softmax (row stats across 16-lane halves) ----------
        float alpha[8];
        #pragma unroll
        for (int r = 0; r < 8; ++r) {
            float mx = fmaxf(fmaxf(s[0][r], s[1][r]), fmaxf(s[2][r], s[3][r]));
            #pragma unroll
            for (int off = 1; off < 16; off <<= 1)
                mx = fmaxf(mx, __shfl_xor(mx, off, 32));
            const float mnew = fmaxf(mrow[r], mx);
            alpha[r] = __expf(mrow[r] - mnew);
            mrow[r] = mnew;
            float sum = 0.0f;
            #pragma unroll
            for (int j = 0; j < 4; ++j) {
                const float p = __expf(s[j][r] - mnew);
                s[j][r] = p;
                sum += p;
            }
            #pragma unroll
            for (int off = 1; off < 16; off <<= 1)
                sum += __shfl_xor(sum, off, 32);
            lrow[r] = lrow[r] * alpha[r] + sum;
        }
        #pragma unroll
        for (int j = 0; j < 4; ++j)
            #pragma unroll
            for (int r = 0; r < 8; ++r) oacc[j][r] *= alpha[r];

        // ---- P: C-layout regs -> LDS (per-wave) -> A-layout frags ------
        #pragma unroll
        for (int j = 0; j < 4; ++j)
            #pragma unroll
            for (int r = 0; r < 8; ++r)
                pw[(r + rowoff) * LDS_KV + j * 16 + l15] = f2bf(s[j][r]);
        __syncthreads();

        // ---- O += P V  (K-dim = key: 0..31 / 32..63) -------------------
        const unsigned short* pbase = &pw[l15 * LDS_KV];
        #pragma unroll
        for (int kk = 0; kk < 2; ++kk) {
            v16bf pa = frag_from(pbase + kk * 32 + hi * 8,
                                 pbase + kk * 32 + 16 + hi * 8);
            #pragma unroll
            for (int j = 0; j < 4; ++j) {
                const unsigned short* vb =
                    &sV[cur][(j * 16 + l15) * LDS_KV + kk * 32 + hi * 8];
                oacc[j] = wmma_bf16(pa, frag_from(vb, vb + 16), oacc[j]);
            }
        }
        __syncthreads();   // next iteration's async issue overwrites `cur`
    }

    // ---- epilogue: O /= l, scatter to [B,T,DIM] fp32 -------------------
    #pragma unroll
    for (int j = 0; j < 4; ++j) {
        #pragma unroll
        for (int r = 0; r < 8; ++r) {
            const int row = qrow0 + r + rowoff;
            const int col = h * HEAD_DIM + j * 16 + l15;
            o[((size_t)b * SEQ + row) * DIM + col] = oacc[j][r] / lrow[r];
        }
    }
}

// ===========================================================================
// Host launch
// ===========================================================================
extern "C" void kernel_launch(void* const* d_in, const int* in_sizes, int n_in,
                              void* d_out, int out_size, void* d_ws, size_t ws_size,
                              hipStream_t stream) {
    const float* x     = (const float*)d_in[0];   // [B,T,DIM] fp32
    const float* w_qkv = (const float*)d_in[1];   // [DIM, 3*DIM] fp32
    const float* w_out = (const float*)d_in[2];   // [DIM, DIM] fp32
    float* out = (float*)d_out;                   // [B,T,DIM] fp32

    const int M = BATCH * SEQ;                    // 8192
    char* ws = (char*)d_ws;
    // workspace layout (bytes)
    unsigned short* qkv = (unsigned short*)(ws);                       // 48 MB
    unsigned short* qb  = (unsigned short*)(ws + (size_t)50331648);    // 16 MB
    unsigned short* kb  = (unsigned short*)(ws + (size_t)67108864);    // 16 MB
    unsigned short* vTb = (unsigned short*)(ws + (size_t)83886080);    // 16 MB
    float*          ob  = (float*)         (ws + (size_t)100663296);   // 32 MB

    // 1) QKV projection: [8192,1024] x [1024,3072] -> bf16 qkv
    gemm_wmma_bf16<true><<<dim3(3 * DIM / 128, M / 128), 256, 0, stream>>>(
        x, w_qkv, qkv, M, 3 * DIM, DIM);

    // 2) RoPE + reshape to heads (+ V transpose)
    rope_reshape<<<dim3((M * DIM) / 256), 256, 0, stream>>>(qkv, qb, kb, vTb);

    // 3) Flash attention: grid(q-tiles, B*H)
    flash_attn<<<dim3(SEQ / 64, BATCH * HEADS), 128, 0, stream>>>(qb, kb, vTb, ob);

    // 4) Output projection: [8192,1024] x [1024,1024] -> fp32 d_out
    gemm_wmma_bf16<false><<<dim3(DIM / 128, M / 128), 256, 0, stream>>>(
        ob, w_out, out, M, DIM, DIM);
}
